// PrototypeNN_76982993814333
// MI455X (gfx1250) — matrix operand order repaired and checked by reading, not compiled, not run on hardware
//
#include <hip/hip_runtime.h>

// N = 131072, D = 256 everywhere, OUT = 2
typedef __attribute__((ext_vector_type(16))) _Float16 v16h;
typedef __attribute__((ext_vector_type(8)))  float    v8f;

union Frag  { v16h h; uint4 q[2]; unsigned int w[8]; };
union H2U   { unsigned int u; _Float16 h[2]; };

static __device__ __forceinline__ unsigned int packh2(float a, float b) {
    H2U x; x.h[0] = (_Float16)a; x.h[1] = (_Float16)b; return x.u;
}

// scheduling-directive helpers (no-ops if builtin unavailable)
#if __has_builtin(__builtin_amdgcn_sched_group_barrier)
#define SGB(mask, size, id) __builtin_amdgcn_sched_group_barrier((mask), (size), (id))
#else
#define SGB(mask, size, id)
#endif
#define SGB_DSREAD 0x100
#define SGB_WMMA   0x008

// ---------------------------------------------------------------------------
// Prep: build pair-packed f16 B-matrices in workspace.
//   Bpk[l][pk][n] (u32) = ( B_l[2pk][n] , B_l[2pk+1][n] )  packed f16x2
//   l=0..2 : B[k][n] = W_l[n][k]        (transposed weights)
//   l=3    : B[k][n] = proto[k][n] / ||proto[k,:]||  (reshape is identity)
// grid = 128 blocks (pk), block = 256 threads (n)
// ---------------------------------------------------------------------------
__global__ __launch_bounds__(256) void prototype_prep_kernel(
    const float* __restrict__ Win, const float* __restrict__ Wh1,
    const float* __restrict__ Wh2, const float* __restrict__ proto,
    unsigned int* __restrict__ Bpk)
{
    const int pk = blockIdx.x;   // 0..127
    const int n  = threadIdx.x;  // 0..255
    const float* Ws[3] = { Win, Wh1, Wh2 };
    for (int l = 0; l < 3; ++l) {
        float lo = Ws[l][n * 256 + 2 * pk];
        float hi = Ws[l][n * 256 + 2 * pk + 1];
        Bpk[l * 32768 + pk * 256 + n] = packh2(lo, hi);
    }
    // prototype rows 2pk, 2pk+1 : row-normalize
    __shared__ float red[256];
    float a = proto[(2 * pk) * 256 + n];
    float b = proto[(2 * pk + 1) * 256 + n];

    red[n] = a * a; __syncthreads();
    for (int s = 128; s > 0; s >>= 1) { if (n < s) red[n] += red[n + s]; __syncthreads(); }
    float sa = red[0]; __syncthreads();
    red[n] = b * b; __syncthreads();
    for (int s = 128; s > 0; s >>= 1) { if (n < s) red[n] += red[n + s]; __syncthreads(); }
    float sb = red[0];

    float ra = 1.0f / sqrtf(sa);
    float rb = 1.0f / sqrtf(sb);
    Bpk[3 * 32768 + pk * 256 + n] = packh2(a * ra, b * rb);
}

// ---------------------------------------------------------------------------
// Fused MLP: 4 waves x 16 rows = 64 rows per workgroup.
// Per layer: K in 8 chunks of 32; weight chunk staged in LDS in a
// fragment-swizzled layout so each lane's B fragment = 2x ds_load_b128.
// sched_group_barriers impose a 3-tile DS-read lookahead over the WMMAs.
// ---------------------------------------------------------------------------
#define LDH  132   // u32 stride of activation rows in LDS (bank-conflict pad)
#define WSTR 12    // u32 per (nt,hi,c) fragment group (8 words + 4 pad)

__global__ __launch_bounds__(128, 1) void prototype_mlp_kernel(
    const float* __restrict__ X,
    const float* __restrict__ b0, const float* __restrict__ b1,
    const float* __restrict__ b2,
    const float* __restrict__ Wout, const float* __restrict__ bout,
    const unsigned int* __restrict__ Bpk,
    float* __restrict__ out)
{
    // swizzled weight chunk: idx = (nt*32 + hi*16 + c)*WSTR + j,  j = pk&7
    __shared__ __align__(16) unsigned int smemW[512 * WSTR];   // 24,576 B
    __shared__ __align__(16) unsigned int smemH[64 * LDH];     // 33,792 B
    __shared__ float smemC[3 * 256 + 512 + 2];                 //  5,128 B

    const int tid  = threadIdx.x;
    const int wave = tid >> 5;
    const int lane = tid & 31;
    const int col  = lane & 15;   // N index inside tile (C/D layout)
    const int hi   = lane >> 4;   // half-wave
    const int rowBase = blockIdx.x * 64;

    // ---- prologue: stage constants + X rows (f32 -> packed f16) -----------
    for (int i = tid; i < 1282; i += 128) {
        float v;
        if      (i <  256) v = b0[i];
        else if (i <  512) v = b1[i - 256];
        else if (i <  768) v = b2[i - 512];
        else if (i < 1280) v = Wout[i - 768];
        else               v = bout[i - 1280];
        smemC[i] = v;
    }
    for (int i = tid; i < 64 * 128; i += 128) {
        int r = i >> 7, p = i & 127;
        const float* src = X + (size_t)(rowBase + r) * 256 + 2 * p;
        smemH[r * LDH + p] = packh2(src[0], src[1]);
    }
    __syncthreads();

    const int hbase = (wave * 16 + col) * LDH;          // A row base (u32)
    const int bbase = (hi * 16 + col) * WSTR;           // B frag base (u32)
    v8f acc[16];

    for (int l = 0; l < 4; ++l) {
        // accumulator init: bias (layers 0-2) or zero (prototype layer)
#pragma unroll
        for (int nt = 0; nt < 16; ++nt) {
            float bv = (l < 3) ? smemC[l * 256 + nt * 16 + col] : 0.0f;
#pragma unroll
            for (int v = 0; v < 8; ++v) acc[nt][v] = bv;
        }

        const unsigned int* Bl = Bpk + l * 32768;
        for (int kc = 0; kc < 8; ++kc) {
            __syncthreads();                     // WAR on previous chunk
            // cooperative swizzled stage: src linear (pk,n) -> frag groups
            const unsigned int* src = Bl + kc * 4096;
            for (int i = tid; i < 4096; i += 128) {
                int pk = i >> 8, n = i & 255;
                int j = pk & 7, h2 = pk >> 3, nt = n >> 4, c = n & 15;
                smemW[(nt * 32 + h2 * 16 + c) * WSTR + j] = src[i];
            }
            if (kc < 7)                          // global_prefetch_b8 next chunk
                __builtin_prefetch((const void*)(Bl + (kc + 1) * 4096 + tid * 32), 0, 1);
            __syncthreads();

            // A fragment: two contiguous b128 loads (p = j + 8*hi)
            Frag afr;
            const unsigned int* ap = &smemH[hbase + kc * 16 + 8 * hi];
            afr.q[0] = *reinterpret_cast<const uint4*>(ap);
            afr.q[1] = *reinterpret_cast<const uint4*>(ap + 4);

#pragma unroll
            for (int nt = 0; nt < 16; ++nt) {
                // B fragment: 8 contiguous words -> two ds_load_b128
                Frag bfr;
                const unsigned int* bp = &smemW[nt * 32 * WSTR + bbase];
                bfr.q[0] = *reinterpret_cast<const uint4*>(bp);
                bfr.q[1] = *reinterpret_cast<const uint4*>(bp + 4);
                acc[nt] = __builtin_amdgcn_wmma_f32_16x16x32_f16(
                    false, afr.h, false, bfr.h, (short)0, acc[nt], false, false);
            }

            // Impose software pipeline: DS reads run 3 tiles ahead of WMMAs.
            // Region has 34 DS reads (2 A + 32 B) and 16 WMMAs:
            //   DSx8 ; { WMMA ; DSx2 } x13 ; WMMAx3
            SGB(SGB_DSREAD, 8, 0);
#pragma unroll
            for (int i = 0; i < 13; ++i) {
                SGB(SGB_WMMA, 1, 0);
                SGB(SGB_DSREAD, 2, 0);
            }
            SGB(SGB_WMMA, 1, 0);
            SGB(SGB_WMMA, 1, 0);
            SGB(SGB_WMMA, 1, 0);
        }

        if (l < 3) {
            // ---- epilogue: relu (+ row L2-normalize after layer 2) --------
#pragma unroll
            for (int nt = 0; nt < 16; ++nt)
#pragma unroll
                for (int v = 0; v < 8; ++v)
                    acc[nt][v] = fmaxf(acc[nt][v], 0.0f);

            float scale[8];
#pragma unroll
            for (int v = 0; v < 8; ++v) scale[v] = 1.0f;
            if (l == 2) {
#pragma unroll
                for (int v = 0; v < 8; ++v) {
                    float ss = 0.0f;
#pragma unroll
                    for (int nt = 0; nt < 16; ++nt) ss += acc[nt][v] * acc[nt][v];
                    ss += __shfl_xor(ss, 1, 32);
                    ss += __shfl_xor(ss, 2, 32);
                    ss += __shfl_xor(ss, 4, 32);
                    ss += __shfl_xor(ss, 8, 32);   // full row sum within half-wave
                    scale[v] = 1.0f / sqrtf(ss);
                }
            }
            // restore next-layer A into per-wave LDS stripe (f16, k-contiguous)
            _Float16* hH = reinterpret_cast<_Float16*>(smemH);
#pragma unroll
            for (int nt = 0; nt < 16; ++nt)
#pragma unroll
                for (int v = 0; v < 8; ++v) {
                    int m = v + 8 * hi;
                    hH[(wave * 16 + m) * (2 * LDH) + nt * 16 + col] =
                        (_Float16)(acc[nt][v] * scale[v]);
                }
        } else {
            // ---- final GEMV: out = inner @ Wout^T + bout ------------------
            float r0[8], r1[8];
#pragma unroll
            for (int v = 0; v < 8; ++v) { r0[v] = 0.0f; r1[v] = 0.0f; }
#pragma unroll
            for (int nt = 0; nt < 16; ++nt) {
                float w0 = smemC[768 +       nt * 16 + col];
                float w1 = smemC[768 + 256 + nt * 16 + col];
#pragma unroll
                for (int v = 0; v < 8; ++v) {
                    r0[v] += acc[nt][v] * w0;
                    r1[v] += acc[nt][v] * w1;
                }
            }
#pragma unroll
            for (int v = 0; v < 8; ++v) {
                r0[v] += __shfl_xor(r0[v], 1, 32);
                r0[v] += __shfl_xor(r0[v], 2, 32);
                r0[v] += __shfl_xor(r0[v], 4, 32);
                r0[v] += __shfl_xor(r0[v], 8, 32);
                r1[v] += __shfl_xor(r1[v], 1, 32);
                r1[v] += __shfl_xor(r1[v], 2, 32);
                r1[v] += __shfl_xor(r1[v], 4, 32);
                r1[v] += __shfl_xor(r1[v], 8, 32);
            }
            if (col == 0) {
                float bo0 = smemC[1280], bo1 = smemC[1281];
#pragma unroll
                for (int v = 0; v < 8; ++v) {
                    size_t row = (size_t)rowBase + wave * 16 + v + 8 * hi;
                    out[row * 2 + 0] = r0[v] + bo0;
                    out[row * 2 + 1] = r1[v] + bo1;
                }
            }
        }
    }
}

extern "C" void kernel_launch(void* const* d_in, const int* in_sizes, int n_in,
                              void* d_out, int out_size, void* d_ws, size_t ws_size,
                              hipStream_t stream) {
    (void)n_in; (void)out_size; (void)ws_size;
    const float* X     = (const float*)d_in[0];
    const float* Win   = (const float*)d_in[1];
    const float* b0    = (const float*)d_in[2];
    const float* Wh1   = (const float*)d_in[3];
    const float* b1    = (const float*)d_in[4];
    const float* Wh2   = (const float*)d_in[5];
    const float* b2    = (const float*)d_in[6];
    const float* proto = (const float*)d_in[7];
    const float* Wout  = (const float*)d_in[8];
    const float* bout  = (const float*)d_in[9];
    unsigned int* Bpk  = (unsigned int*)d_ws;   // 4 * 32768 u32 = 512 KB
    float* out         = (float*)d_out;

    const int N = in_sizes[0] / 256;            // 131072

    prototype_prep_kernel<<<dim3(128), dim3(256), 0, stream>>>(
        Win, Wh1, Wh2, proto, Bpk);
    prototype_mlp_kernel<<<dim3(N / 64), dim3(128), 0, stream>>>(
        X, b0, b1, b2, Wout, bout, Bpk, out);
}
